// GAT_82386062671996
// MI455X (gfx1250) — matrix operand order repaired and checked by reading, not compiled, not run on hardware
//
#include <hip/hip_runtime.h>
#include <hip/hip_bf16.h>

typedef _Float16 v16h __attribute__((ext_vector_type(16)));
typedef float    v8f  __attribute__((ext_vector_type(8)));

#define NEG_SLOPE 0.2f

// ---------------- helpers ----------------
__device__ inline unsigned pack2h(float a, float b) {
    union { _Float16 h[2]; unsigned u; } p;
    p.h[0] = (_Float16)a; p.h[1] = (_Float16)b;
    return p.u;
}
// order-preserving float<->uint encoding for atomicMax on signed floats
__device__ inline unsigned fenc(float f) {
    unsigned u = __float_as_uint(f);
    return (u & 0x80000000u) ? ~u : (u | 0x80000000u);
}
__device__ inline float fdec(unsigned u) {
    unsigned b = (u & 0x80000000u) ? (u & 0x7FFFFFFFu) : ~u;
    return __uint_as_float(b);
}

// 32-bit LDS (addrspace 3) byte offset of a shared-memory pointer
__device__ inline unsigned lds_addr32(const void* p) {
    return (unsigned)(uintptr_t)(__attribute__((address_space(3))) const void*)p;
}

// gfx1250 async copy: global memory -> LDS, 16 bytes per lane (ASYNCcnt-tracked)
__device__ inline void async_g2lds_b128(unsigned lds_off, unsigned long long gaddr) {
    asm volatile("global_load_async_to_lds_b128 %0, %1, off"
                 :: "v"(lds_off), "v"(gaddr)
                 : "memory");
}
__device__ inline void wait_asynccnt0() {
    asm volatile("s_wait_asynccnt 0" ::: "memory");
}

// ---------------- WMMA GEMM: H[N x Wcols] = X[N x Cin] * W[Cin x Wcols] ----
// 256 threads = 8 wave32. Block tile: 128 rows x (NB*16) cols, K in chunks of 32.
// X tile staged via global_load_async_to_lds_b128 (raw f32), converted to f16
// packs during fragment build; W tile staged synchronously (needs transpose).
template <int NB>
__global__ __launch_bounds__(256) void gat_gemm_wmma(
    const float* __restrict__ X, const float* __restrict__ W,
    float* __restrict__ H, int N, int Cin, int Wcols, int outStride)
{
    constexpr int COUT = NB * 16;
    constexpr int ASTR = 36;               // f32 row stride: 144B (16B-aligned, bank-spread)
    __shared__ float    Asf[128 * ASTR];   // [row][k] raw f32 (async-filled)
    __shared__ _Float16 Bs[COUT * 34];     // [n][k] (W transposed) f16, stride 34

    const int tid  = threadIdx.x;
    const int wv   = tid >> 5;
    const int lane = tid & 31;
    const int hl   = lane >> 4;            // lane half (0: lanes 0-15, 1: 16-31)
    const int l16  = lane & 15;
    const int blockRow = blockIdx.x * 128;

    v8f acc[NB];
#pragma unroll
    for (int nb = 0; nb < NB; ++nb)
        acc[nb] = (v8f){0.f,0.f,0.f,0.f,0.f,0.f,0.f,0.f};

    const int nK = Cin >> 5;
    for (int kc = 0; kc < nK; ++kc) {
        // ---- async-stage X tile (128 x 32 f32): 1024 x 16B chunks ----
#pragma unroll
        for (int j = 0; j < 4; ++j) {
            int i    = tid + j * 256;
            int row  = i >> 3;                       // 8 x float4 per row
            int c4   = (i & 7) * 4;
            int rowG = blockRow + row;
            rowG = (rowG < N) ? rowG : (N - 1);      // clamp: keep EXEC full
            unsigned long long g =
                (unsigned long long)(uintptr_t)(X + (size_t)rowG * Cin + kc * 32 + c4);
            async_g2lds_b128(lds_addr32(&Asf[row * ASTR + c4]), g);
        }
        // ---- stage W tile transposed: Bs[n][k] = (f16)W[kc*32+k][n] ----
        for (int idx = tid; idx < 32 * COUT; idx += 256) {
            int k = idx / COUT;
            int n = idx % COUT;
            float wval = (n < Wcols) ? W[(size_t)(kc * 32 + k) * Wcols + n] : 0.f;
            Bs[n * 34 + k] = (_Float16)wval;
        }
        wait_asynccnt0();
        __syncthreads();

        // ---- A fragment (16x32 f16): lane-half K offset 8 (ISA 7.12.2) ----
        union { v16h v; unsigned u[8]; } af;
        const int mloc = wv * 16 + l16;
#pragma unroll
        for (int i = 0; i < 8; ++i) {
            int kb = (i < 4) ? (i * 2 + hl * 8) : (16 + (i - 4) * 2 + hl * 8);
            float2 f = *(const float2*)&Asf[mloc * ASTR + kb];   // ds_load_b64
            af.u[i] = pack2h(f.x, f.y);                          // v_cvt_pk_f16_f32
        }
        // ---- B fragments (32x16 f16): lane-half K offset 8 ----
#pragma unroll
        for (int nb = 0; nb < NB; ++nb) {
            union { v16h v; unsigned u[8]; } bf;
            const int nloc = nb * 16 + l16;
#pragma unroll
            for (int i = 0; i < 8; ++i)
                bf.u[i] = *(const unsigned*)&Bs[nloc * 34 + ((i < 4) ? (i * 2 + hl * 8) : (16 + (i - 4) * 2 + hl * 8))];
            acc[nb] = __builtin_amdgcn_wmma_f32_16x16x32_f16(
                false, af.v, false, bf.v, (short)0, acc[nb], false, false);
        }
        __syncthreads();
    }

    // ---- epilogue: D layout M = r + 8*half, N = lane&15 ----
#pragma unroll
    for (int nb = 0; nb < NB; ++nb) {
        int n = nb * 16 + l16;
        if (n >= Wcols) continue;
#pragma unroll
        for (int r = 0; r < 8; ++r) {
            int m = blockRow + wv * 16 + r + hl * 8;
            if (m < N) H[(size_t)m * outStride + n] = acc[nb][r];
        }
    }
}

// ---------------- per-node attention logits ----------------
__global__ void gat_logits(const float* __restrict__ H,
                           const float* __restrict__ a_src,
                           const float* __restrict__ a_dst,
                           float* __restrict__ alsrc, float* __restrict__ aldst,
                           int N, int heads, int D)
{
    int n = blockIdx.x * blockDim.x + threadIdx.x;
    if (n >= N) return;
    const float* hp = H + (size_t)n * heads * D;
    for (int hd = 0; hd < heads; ++hd) {
        float s1 = 0.f, s2 = 0.f;
        for (int d = 0; d < D; ++d) {
            float hv = hp[hd * D + d];
            s1 += hv * a_src[hd * D + d];
            s2 += hv * a_dst[hd * D + d];
        }
        alsrc[n * heads + hd] = s1;
        aldst[n * heads + hd] = s2;
    }
}

// ---------------- per-edge score + segment max ----------------
__global__ void gat_edge_score(const int* __restrict__ src, const int* __restrict__ dst,
                               const float* __restrict__ alsrc, const float* __restrict__ aldst,
                               float* __restrict__ ebuf, unsigned* __restrict__ menc,
                               int E, int heads)
{
    int t = blockIdx.x * blockDim.x + threadIdx.x;
    if (t >= E * heads) return;
    int e = t / heads, hd = t - e * heads;
    int s = src[e], d = dst[e];
    float v = alsrc[s * heads + hd] + aldst[d * heads + hd];
    v = (v > 0.f) ? v : NEG_SLOPE * v;           // leaky relu
    ebuf[t] = v;
    atomicMax(&menc[d * heads + hd], fenc(v));
}

// ---------------- per-edge exp + segment sum ----------------
__global__ void gat_edge_exp(const int* __restrict__ dst,
                             float* __restrict__ ebuf,            // in: score, out: exp weight
                             const unsigned* __restrict__ menc,
                             float* __restrict__ denom,
                             int E, int heads)
{
    int t = blockIdx.x * blockDim.x + threadIdx.x;
    if (t >= E * heads) return;
    int e = t / heads, hd = t - e * heads;
    int d = dst[e];
    float m = fdec(menc[d * heads + hd]);
    float w = __expf(ebuf[t] - m);
    ebuf[t] = w;
    atomicAdd(&denom[d * heads + hd], w);
}

// ---------------- wave-per-edge weighted gather / scatter-add ----------------
__global__ void gat_aggregate(const int* __restrict__ src, const int* __restrict__ dst,
                              const float* __restrict__ w, const float* __restrict__ denom,
                              const float* __restrict__ H, float* __restrict__ out,
                              int E, int heads, int D)
{
    int t = blockIdx.x * blockDim.x + threadIdx.x;
    int e = t >> 5;
    int lane = t & 31;
    if (e >= E) return;
    int s = src[e], d = dst[e];
    int C = heads * D;
    for (int c = lane; c < C; c += 32) {
        int hd = c / D;
        float alpha = w[(size_t)e * heads + hd] / denom[(size_t)d * heads + hd];
        atomicAdd(&out[(size_t)d * C + c], alpha * H[(size_t)s * C + c]);
    }
}

// ---------------- bias (+ optional relu) ----------------
__global__ void gat_bias_act(float* __restrict__ buf, const float* __restrict__ bias,
                             size_t total, int C, int doRelu)
{
    size_t idx = (size_t)blockIdx.x * blockDim.x + threadIdx.x;
    if (idx >= total) return;
    float v = buf[idx] + bias[idx % C];
    if (doRelu) v = fmaxf(v, 0.f);
    buf[idx] = v;
}

__global__ void fill_u32(unsigned* __restrict__ p, unsigned v, size_t n)
{
    size_t idx = (size_t)blockIdx.x * blockDim.x + threadIdx.x;
    if (idx < n) p[idx] = v;
}

// ---------------- host orchestration ----------------
static inline int cdiv(long long a, long long b) { return (int)((a + b - 1) / b); }

extern "C" void kernel_launch(void* const* d_in, const int* in_sizes, int n_in,
                              void* d_out, int out_size, void* d_ws, size_t ws_size,
                              hipStream_t stream)
{
    const float* x      = (const float*)d_in[0];
    const int*   eidx   = (const int*)d_in[1];
    const float* W1     = (const float*)d_in[2];
    const float* asrc1  = (const float*)d_in[3];
    const float* adst1  = (const float*)d_in[4];
    const float* b1     = (const float*)d_in[5];
    const float* W2     = (const float*)d_in[6];
    const float* asrc2  = (const float*)d_in[7];
    const float* adst2  = (const float*)d_in[8];
    const float* b2     = (const float*)d_in[9];
    const float* W3     = (const float*)d_in[10];
    const float* asrc3  = (const float*)d_in[11];
    const float* adst3  = (const float*)d_in[12];
    const float* b3     = (const float*)d_in[13];

    const int IN = 128, HID = 128, HEADS = 4, D = 32;
    const int N = in_sizes[0] / IN;
    const int E = in_sizes[1] / 2;
    const int* src = eidx;
    const int* dst = eidx + E;

    // workspace carve-up (floats)
    float* ws = (float*)d_ws;
    float*    hbuf   = ws;                       // N*128
    float*    aggbuf = hbuf   + (size_t)N * 128; // N*128
    float*    alsrc  = aggbuf + (size_t)N * 128; // N*4
    float*    aldst  = alsrc  + (size_t)N * 4;   // N*4
    float*    denom  = aldst  + (size_t)N * 4;   // N*4
    unsigned* menc   = (unsigned*)(denom + (size_t)N * 4); // N*4
    float*    ebuf   = (float*)(menc + (size_t)N * 4);     // E*4

    const int T = 256;
    const int gGemm  = cdiv(N, 128);
    const int gNode  = cdiv(N, T);

    // ======================= layer 1 & 2 (heads=4, D=32) =======================
    const float* xin = x;
    for (int layer = 0; layer < 2; ++layer) {
        const float* W  = (layer == 0) ? W1 : W2;
        const float* as = (layer == 0) ? asrc1 : asrc2;
        const float* ad = (layer == 0) ? adst1 : adst2;
        const float* bb = (layer == 0) ? b1 : b2;

        gat_gemm_wmma<8><<<gGemm, T, 0, stream>>>(xin, W, hbuf, N, 128, HID, HID);
        gat_logits<<<gNode, T, 0, stream>>>(hbuf, as, ad, alsrc, aldst, N, HEADS, D);

        fill_u32<<<cdiv((long long)N * 4, T), T, 0, stream>>>(menc, 0u, (size_t)N * 4);
        fill_u32<<<cdiv((long long)N * 4, T), T, 0, stream>>>((unsigned*)denom, 0u, (size_t)N * 4);
        fill_u32<<<cdiv((long long)N * 128, T), T, 0, stream>>>((unsigned*)aggbuf, 0u, (size_t)N * 128);

        gat_edge_score<<<cdiv((long long)E * HEADS, T), T, 0, stream>>>(
            src, dst, alsrc, aldst, ebuf, menc, E, HEADS);
        gat_edge_exp<<<cdiv((long long)E * HEADS, T), T, 0, stream>>>(
            dst, ebuf, menc, denom, E, HEADS);
        gat_aggregate<<<cdiv((long long)E * 32, T), T, 0, stream>>>(
            src, dst, ebuf, denom, hbuf, aggbuf, E, HEADS, D);
        gat_bias_act<<<cdiv((long long)N * 128, T), T, 0, stream>>>(
            aggbuf, bb, (size_t)N * 128, HID, 1);

        xin = aggbuf;
    }

    // ======================= layer 3 (heads=1, D=2) =======================
    float* out = (float*)d_out;
    gat_gemm_wmma<1><<<gGemm, T, 0, stream>>>(xin, W3, hbuf, N, 128, 2, 2);
    gat_logits<<<gNode, T, 0, stream>>>(hbuf, asrc3, adst3, alsrc, aldst, N, 1, 2);

    fill_u32<<<cdiv(N, T), T, 0, stream>>>(menc, 0u, (size_t)N);
    fill_u32<<<cdiv(N, T), T, 0, stream>>>((unsigned*)denom, 0u, (size_t)N);
    fill_u32<<<cdiv((long long)N * 2, T), T, 0, stream>>>((unsigned*)out, 0u, (size_t)N * 2);

    gat_edge_score<<<cdiv(E, T), T, 0, stream>>>(src, dst, alsrc, aldst, ebuf, menc, E, 1);
    gat_edge_exp<<<cdiv(E, T), T, 0, stream>>>(dst, ebuf, menc, denom, E, 1);
    gat_aggregate<<<cdiv((long long)E * 32, T), T, 0, stream>>>(
        src, dst, ebuf, denom, hbuf, out, E, 1, 2);
    gat_bias_act<<<cdiv((long long)N * 2, T), T, 0, stream>>>(out, b3, (size_t)N * 2, 2, 0);
}
